// KEANNKG_13417477833154
// MI455X (gfx1250) — compile-verified, matrix operand order checked by reading, not run on hardware
//
#include <hip/hip_runtime.h>
#include <hip/hip_bf16.h>
#include <cstdint>
#include <cstddef>

// ---------------------------------------------------------------------------
// Problem constants (from the reference)
// ---------------------------------------------------------------------------
constexpr int B  = 64;
constexpr int L  = 80;
constexpr int E  = 256;
constexpr int H  = 256;
constexpr int H4 = 4 * H;        // 1024
constexpr int N  = B * L;        // 5120
constexpr int TH = 2 * H;        // 512
constexpr int D  = 4 * L * H;    // 81920
constexpr int C  = 4;
constexpr int DT = 100;
constexpr int M2 = 2 * N;        // 10240 stacked rows (e1 then e2)
constexpr float EPS = 1e-5f;

// ---------------------------------------------------------------------------
// WMMA types (gfx1250, wave32)
// ---------------------------------------------------------------------------
typedef __bf16 v16bf __attribute__((ext_vector_type(16)));
typedef float  v8f   __attribute__((ext_vector_type(8)));

static __device__ __forceinline__ v8f wmma_bf16(v16bf a, v16bf b, v8f c) {
  // D = A(16x32) * B(32x16) + C(16x16 f32)
  return __builtin_amdgcn_wmma_f32_16x16x32_bf16(
      /*neg_a=*/false, a, /*neg_b=*/false, b,
      /*c_mod=*/(short)0, c, /*reuse_a=*/false, /*reuse_b=*/false);
}

static __device__ __forceinline__ float sigmoidf_(float x) {
  return 1.0f / (1.0f + expf(-x));
}

// Block reductions (256 threads)
static __device__ __forceinline__ float blk_sum(float v, float* sd) {
  int t = threadIdx.x; sd[t] = v; __syncthreads();
#pragma unroll
  for (int s = 128; s > 0; s >>= 1) { if (t < s) sd[t] += sd[t + s]; __syncthreads(); }
  float r = sd[0]; __syncthreads(); return r;
}
static __device__ __forceinline__ float blk_max(float v, float* sd) {
  int t = threadIdx.x; sd[t] = v; __syncthreads();
#pragma unroll
  for (int s = 128; s > 0; s >>= 1) { if (t < s) sd[t] = fmaxf(sd[t], sd[t + s]); __syncthreads(); }
  float r = sd[0]; __syncthreads(); return r;
}

// ---------------------------------------------------------------------------
// Small prep kernels
// ---------------------------------------------------------------------------
__global__ void k_rel(const float* __restrict__ tE1, const float* __restrict__ tE2,
                      float* __restrict__ rel) {
  __shared__ float m1s[L], m2s[L];
  int tid = threadIdx.x;
  if (tid < L) {
    float s = 0.f;
    for (int j = 0; j < DT; ++j) s += tE1[tid * DT + j];
    m1s[tid] = s * (1.0f / DT);
  } else if (tid >= 128 && tid < 128 + L) {
    int r = tid - 128; float s = 0.f;
    for (int j = 0; j < DT; ++j) s += tE2[r * DT + j];
    m2s[r] = s * (1.0f / DT);
  }
  __syncthreads();
  for (int idx = tid; idx < L * L; idx += 256) {
    int i = idx / L, j = idx % L;
    rel[idx] = tanhf(m1s[i] - m2s[j]);
  }
}

__global__ void k_bias_sum(const float* __restrict__ a, const float* __restrict__ b,
                           float* __restrict__ o, int n) {
  int i = blockIdx.x * 256 + threadIdx.x;
  if (i < n) o[i] = a[i] + b[i];
}

__global__ void k_cvt_bf16(const float* __restrict__ s, __bf16* __restrict__ d, int n) {
  int i = blockIdx.x * 256 + threadIdx.x;
  if (i < n) d[i] = (__bf16)s[i];
}

// Rt[n][k] = R[k][n]  (R is (2E,2E) row-major; we need B-operand layout (N x K))
__global__ void k_rt_cvt(const float* __restrict__ R, __bf16* __restrict__ Rt) {
  int gid = blockIdx.x * 256 + threadIdx.x;   // n*512 + k
  int n = gid >> 9, k = gid & 511;
  Rt[gid] = (__bf16)R[(size_t)k * TH + n];
}

// estack[(m*L + t)*E + k] = bf16(table[arg[b][t]][k]); m<64 -> arg1, else arg2
__global__ void k_gather(const int* __restrict__ a1, const int* __restrict__ a2,
                         const float* __restrict__ table, __bf16* __restrict__ estack) {
  size_t gid = (size_t)blockIdx.x * 256 + threadIdx.x;
  int row = (int)(gid >> 8), k = (int)(gid & 255);
  int m = row / L, t = row % L;
  const int* a = (m < B) ? a1 : a2;
  int b = (m < B) ? m : m - B;
  int w = a[b * L + t];
  estack[gid] = (__bf16)table[(size_t)w * E + k];
}

// ---------------------------------------------------------------------------
// Generic WMMA GEMM:  Dst(MxN,f32) = A(MxK,bf16) * Bw(NxK,bf16)^T (+bias)
// grid = (N/128, M/64), block = 256 (8 waves); wave tile = 32x32.
// Used for the three big one-shot GEMMs (xg_f, xg_b, t1R).
// ---------------------------------------------------------------------------
__global__ __launch_bounds__(256)
void k_gemm_bf16(const __bf16* __restrict__ A, int lda,
                 const __bf16* __restrict__ Bw, int ldb,
                 float* __restrict__ Dst, int ldd,
                 const float* __restrict__ bias, int K) {
  const int lane = threadIdx.x & 31;
  const int w    = threadIdx.x >> 5;
  const int m0 = blockIdx.y * 64  + (w & 1) * 32;
  const int n0 = blockIdx.x * 128 + (w >> 1) * 32;
  const int la = lane & 15, lh = lane >> 4;

  const __bf16* Ar0 = A  + (size_t)(m0 + la) * lda + lh * 16;
  const __bf16* Ar1 = Ar0 + (size_t)16 * lda;
  const __bf16* Br0 = Bw + (size_t)(n0 + la) * ldb + lh * 16;
  const __bf16* Br1 = Br0 + (size_t)16 * ldb;

  v8f acc[2][2] = {};
  for (int k0 = 0; k0 < K; k0 += 32) {
    v16bf a0 = *(const v16bf*)(Ar0 + k0);
    v16bf a1 = *(const v16bf*)(Ar1 + k0);
    v16bf b0 = *(const v16bf*)(Br0 + k0);
    v16bf b1 = *(const v16bf*)(Br1 + k0);
    acc[0][0] = wmma_bf16(a0, b0, acc[0][0]);
    acc[0][1] = wmma_bf16(a0, b1, acc[0][1]);
    acc[1][0] = wmma_bf16(a1, b0, acc[1][0]);
    acc[1][1] = wmma_bf16(a1, b1, acc[1][1]);
  }

#pragma unroll
  for (int ti = 0; ti < 2; ++ti)
#pragma unroll
    for (int tj = 0; tj < 2; ++tj) {
      int rbase = m0 + ti * 16 + lh * 8;
      int cg    = n0 + tj * 16 + la;
      float bv = bias ? bias[cg] : 0.0f;
#pragma unroll
      for (int v = 0; v < 8; ++v) {
        int r = rbase + v;
        Dst[(size_t)r * ldd + cg] = acc[ti][tj][v] + bv;
      }
    }
}

// ---------------------------------------------------------------------------
// Fused "last" GEMM: last = tanh(t1R * t2^T) + blockdiag(rel).  M=N=5120,K=512
// Wave tile 32x64 (8 accumulators, 8 wmma per 6 fragment loads);
// block tile 64x256; grid = (20, 80).
// ---------------------------------------------------------------------------
__global__ __launch_bounds__(256)
void k_gemm_last(const __bf16* __restrict__ A, const __bf16* __restrict__ Bw,
                 float* __restrict__ last, const float* __restrict__ rel, int K) {
  const int lane = threadIdx.x & 31;
  const int w    = threadIdx.x >> 5;
  const int m0 = blockIdx.y * 64  + (w & 1) * 32;
  const int n0 = blockIdx.x * 256 + (w >> 1) * 64;
  const int la = lane & 15, lh = lane >> 4;

  const __bf16* Ar0 = A  + (size_t)(m0 + la) * TH + lh * 16;
  const __bf16* Ar1 = Ar0 + (size_t)16 * TH;
  const __bf16* Br[4];
#pragma unroll
  for (int tb = 0; tb < 4; ++tb)
    Br[tb] = Bw + (size_t)(n0 + tb * 16 + la) * TH + lh * 16;

  v8f acc[2][4] = {};
  for (int k0 = 0; k0 < K; k0 += 32) {
    v16bf a0 = *(const v16bf*)(Ar0 + k0);
    v16bf a1 = *(const v16bf*)(Ar1 + k0);
#pragma unroll
    for (int tb = 0; tb < 4; ++tb) {
      v16bf b = *(const v16bf*)(Br[tb] + k0);
      acc[0][tb] = wmma_bf16(a0, b, acc[0][tb]);
      acc[1][tb] = wmma_bf16(a1, b, acc[1][tb]);
    }
  }

#pragma unroll
  for (int ti = 0; ti < 2; ++ti)
#pragma unroll
    for (int tj = 0; tj < 4; ++tj) {
      int rbase = m0 + ti * 16 + lh * 8;
      int jg    = n0 + tj * 16 + la;
      int jb = jg / L, jr = jg % L;
#pragma unroll
      for (int v = 0; v < 8; ++v) {
        int ig = rbase + v;
        float x = tanhf(acc[ti][tj][v]);
        if (ig / L == jb) x += rel[(ig % L) * L + jr];
        last[(size_t)ig * N + jg] = x;
      }
    }
}

// ---------------------------------------------------------------------------
// Fused LSTM step: gates = xg[:,t] + h*Whh^T (WMMA), staged in LDS, then the
// i/f/g/o pointwise update — one kernel per time step (halves chain depth).
// grid = 8 blocks (16-row M tiles over the stacked M=128), block = 256.
// Each wave computes a 16x128 gate slice: 8 accumulators, K=256.
// h is ping-pong buffered across steps (blocks read and write only their own 16 rows).
// ---------------------------------------------------------------------------
__global__ __launch_bounds__(256)
void k_lstm_step(const __bf16* __restrict__ hbf_in, const __bf16* __restrict__ Whh,
                 const float* __restrict__ xg_t, float* __restrict__ cbuf,
                 __bf16* __restrict__ hbf_out, float* __restrict__ t1buf,
                 float* __restrict__ t2buf, int t_store, int col_off) {
  __shared__ float sg[16 * H4];   // 64 KB gate tile
  const int lane = threadIdx.x & 31;
  const int w    = threadIdx.x >> 5;
  const int m0 = blockIdx.x * 16;
  const int n0 = w * 128;
  const int la = lane & 15, lh = lane >> 4;

  const __bf16* Ar = hbf_in + (size_t)(m0 + la) * H + lh * 16;
  const __bf16* Br[8];
#pragma unroll
  for (int tb = 0; tb < 8; ++tb)
    Br[tb] = Whh + (size_t)(n0 + tb * 16 + la) * H + lh * 16;

  v8f acc[8] = {};
  for (int k0 = 0; k0 < H; k0 += 32) {
    v16bf a = *(const v16bf*)(Ar + k0);
#pragma unroll
    for (int tb = 0; tb < 8; ++tb) {
      v16bf b = *(const v16bf*)(Br[tb] + k0);
      acc[tb] = wmma_bf16(a, b, acc[tb]);
    }
  }

  // epilogue: add xg (row stride L*H4) and stage gates in LDS
#pragma unroll
  for (int tb = 0; tb < 8; ++tb) {
    int cg = n0 + tb * 16 + la;
#pragma unroll
    for (int v = 0; v < 8; ++v) {
      int r = lh * 8 + v;                       // 0..15 local row
      float x = acc[tb][v] + xg_t[(size_t)(m0 + r) * (L * H4) + cg];
      sg[r * H4 + cg] = x;
    }
  }
  __syncthreads();

  // pointwise LSTM cell update for the 16x256 rows owned by this block
  for (int idx = threadIdx.x; idx < 16 * H; idx += 256) {
    int ml = idx >> 8, hh = idx & 255;
    const float* g = sg + ml * H4;
    float gi = sigmoidf_(g[hh]);
    float gf = sigmoidf_(g[H + hh]);
    float gg = tanhf(g[2 * H + hh]);
    float go = sigmoidf_(g[3 * H + hh]);
    int m = m0 + ml;
    int gidx = m * H + hh;
    float cn = gf * cbuf[gidx] + gi * gg;
    cbuf[gidx] = cn;
    float h = go * tanhf(cn);
    hbf_out[gidx] = (__bf16)h;
    float* obuf = (m < B) ? t1buf : t2buf;
    int b = m & (B - 1);
    obuf[((size_t)(b * L + t_store)) * TH + col_off + hh] = h;
  }
}

// ---------------------------------------------------------------------------
// Softmax statistics / sf reductions over last (5120x5120)
// ---------------------------------------------------------------------------
__global__ void k_rowstat(const float* __restrict__ last, float* __restrict__ rmax,
                          float* __restrict__ rsum) {
  __shared__ float sd[256];
  int i = blockIdx.x;
  const float* row = last + (size_t)i * N;
  float m = -3.4e38f;
  for (int j = threadIdx.x; j < N; j += 256) m = fmaxf(m, row[j]);
  m = blk_max(m, sd);
  float s = 0.f;
  for (int j = threadIdx.x; j < N; j += 256) s += expf(row[j] - m);
  s = blk_sum(s, sd);
  if (threadIdx.x == 0) { rmax[i] = m; rsum[i] = s; }
}

__global__ void k_colstat(const float* __restrict__ last, float* __restrict__ cmax,
                          float* __restrict__ csum) {
  __shared__ float sd[256];
  int j = blockIdx.x;
  float m = -3.4e38f;
  for (int i = threadIdx.x; i < N; i += 256) m = fmaxf(m, last[(size_t)i * N + j]);
  m = blk_max(m, sd);
  float s = 0.f;
  for (int i = threadIdx.x; i < N; i += 256) s += expf(last[(size_t)i * N + j] - m);
  s = blk_sum(s, sd);
  if (threadIdx.x == 0) { cmax[j] = m; csum[j] = s; }
}

// sf1[j] = mean_i softmax_row(i)[j]
__global__ void k_sf1(const float* __restrict__ last, const float* __restrict__ rmax,
                      const float* __restrict__ rsum, float* __restrict__ sf1) {
  __shared__ float sd[256];
  int j = blockIdx.x;
  float s = 0.f;
  for (int i = threadIdx.x; i < N; i += 256)
    s += expf(last[(size_t)i * N + j] - rmax[i]) / rsum[i];
  s = blk_sum(s, sd);
  if (threadIdx.x == 0) sf1[j] = s * (1.0f / N);
}

// sf2[i] = mean_j softmax_col(j)[i]
__global__ void k_sf2(const float* __restrict__ last, const float* __restrict__ cmax,
                      const float* __restrict__ csum, float* __restrict__ sf2) {
  __shared__ float sd[256];
  int i = blockIdx.x;
  const float* row = last + (size_t)i * N;
  float s = 0.f;
  for (int j = threadIdx.x; j < N; j += 256)
    s += expf(row[j] - cmax[j]) / csum[j];
  s = blk_sum(s, sd);
  if (threadIdx.x == 0) sf2[i] = s * (1.0f / N);
}

// ---------------------------------------------------------------------------
// out(b,d) recomputed on the fly; BatchNorm stats; fused BN+FC; log_softmax
// ---------------------------------------------------------------------------
static __device__ __forceinline__ float out_val(int b, int d,
    const float* __restrict__ t1, const float* __restrict__ t2,
    const float* __restrict__ sf1, const float* __restrict__ sf2) {
  int l2 = d >> 9, hc = d & 511;
  if (l2 < L) { int n = b * L + l2;        return t1[(size_t)n * TH + hc] * sf2[n]; }
  else        { int n = b * L + (l2 - L);  return t2[(size_t)n * TH + hc] * sf1[n]; }
}

__global__ void k_bnstat(const float* __restrict__ t1, const float* __restrict__ t2,
                         const float* __restrict__ sf1, const float* __restrict__ sf2,
                         float* __restrict__ mu, float* __restrict__ var) {
  int d = blockIdx.x * 256 + threadIdx.x;
  if (d >= D) return;
  float s = 0.f, s2 = 0.f;
  for (int b = 0; b < B; ++b) {
    float v = out_val(b, d, t1, t2, sf1, sf2);
    s += v; s2 += v * v;
  }
  float m = s * (1.0f / B);
  mu[d] = m;
  var[d] = s2 * (1.0f / B) - m * m;   // biased variance (training-mode BN)
}

__global__ void k_fc(const float* __restrict__ t1, const float* __restrict__ t2,
                     const float* __restrict__ sf1, const float* __restrict__ sf2,
                     const float* __restrict__ mu, const float* __restrict__ var,
                     const float* __restrict__ gamma, const float* __restrict__ beta,
                     const float* __restrict__ fcW, const float* __restrict__ fcb,
                     float* __restrict__ logits) {
  __shared__ float sd[256];
  int b = blockIdx.x >> 2, c = blockIdx.x & 3;
  float p = 0.f;
  for (int d = threadIdx.x; d < D; d += 256) {
    float v  = out_val(b, d, t1, t2, sf1, sf2);
    float xn = (v - mu[d]) * rsqrtf(var[d] + EPS) * gamma[d] + beta[d];
    p += xn * fcW[(size_t)c * D + d];
  }
  p = blk_sum(p, sd);
  if (threadIdx.x == 0) logits[b * C + c] = p + fcb[c];
}

__global__ void k_logsoftmax(const float* __restrict__ logits, float* __restrict__ out) {
  int b = threadIdx.x;
  if (b >= B) return;
  float x0 = logits[b * C + 0], x1 = logits[b * C + 1];
  float x2 = logits[b * C + 2], x3 = logits[b * C + 3];
  float m = fmaxf(fmaxf(x0, x1), fmaxf(x2, x3));
  float lse = logf(expf(x0 - m) + expf(x1 - m) + expf(x2 - m) + expf(x3 - m));
  out[b * C + 0] = x0 - m - lse;
  out[b * C + 1] = x1 - m - lse;
  out[b * C + 2] = x2 - m - lse;
  out[b * C + 3] = x3 - m - lse;
}

// ---------------------------------------------------------------------------
// Host launcher
// ---------------------------------------------------------------------------
extern "C" void kernel_launch(void* const* d_in, const int* in_sizes, int n_in,
                              void* d_out, int out_size, void* d_ws, size_t ws_size,
                              hipStream_t stream) {
  (void)in_sizes; (void)n_in; (void)out_size;
  const int*   arg1  = (const int*)d_in[0];
  const int*   arg2  = (const int*)d_in[1];
  const float* tE1   = (const float*)d_in[3];
  const float* tE2   = (const float*)d_in[4];
  const float* table = (const float*)d_in[5];
  const float* WihF  = (const float*)d_in[6];
  const float* WhhF  = (const float*)d_in[7];
  const float* bihF  = (const float*)d_in[8];
  const float* bhhF  = (const float*)d_in[9];
  const float* WihB  = (const float*)d_in[10];
  const float* WhhB  = (const float*)d_in[11];
  const float* bihB  = (const float*)d_in[12];
  const float* bhhB  = (const float*)d_in[13];
  const float* Rm    = (const float*)d_in[14];
  const float* gamma = (const float*)d_in[15];
  const float* beta  = (const float*)d_in[16];
  const float* fcW   = (const float*)d_in[17];
  const float* fcb   = (const float*)d_in[18];
  float* outp = (float*)d_out;

  // Workspace layout (256B aligned slices)
  char* ws = (char*)d_ws;
  size_t off = 0;
  auto take = [&](size_t bytes) { size_t o = off; off += (bytes + 255) & ~(size_t)255; return o; };

  float*  rel    = (float*)(ws + take(L * L * 4));
  float*  bsumF  = (float*)(ws + take(H4 * 4));
  float*  bsumB  = (float*)(ws + take(H4 * 4));
  __bf16* WihFb  = (__bf16*)(ws + take((size_t)H4 * E * 2));
  __bf16* WihBb  = (__bf16*)(ws + take((size_t)H4 * E * 2));
  __bf16* WhhFb  = (__bf16*)(ws + take((size_t)H4 * H * 2));
  __bf16* WhhBb  = (__bf16*)(ws + take((size_t)H4 * H * 2));
  __bf16* Rtb    = (__bf16*)(ws + take((size_t)TH * TH * 2));
  __bf16* estack = (__bf16*)(ws + take((size_t)M2 * E * 2));
  float*  t1buf  = (float*)(ws + take((size_t)N * TH * 4));
  float*  t2buf  = (float*)(ws + take((size_t)N * TH * 4));
  __bf16* t1bf   = (__bf16*)(ws + take((size_t)N * TH * 2));
  __bf16* t2bf   = (__bf16*)(ws + take((size_t)N * TH * 2));
  float*  t1R    = (float*)(ws + take((size_t)N * TH * 4));
  __bf16* t1Rbf  = (__bf16*)(ws + take((size_t)N * TH * 2));
  __bf16* hbfA   = (__bf16*)(ws + take((size_t)2 * B * H * 2));
  __bf16* hbfB   = (__bf16*)(ws + take((size_t)2 * B * H * 2));
  float*  cbuf   = (float*)(ws + take((size_t)2 * B * H * 4));
  float*  rmax   = (float*)(ws + take(N * 4));
  float*  rsum   = (float*)(ws + take(N * 4));
  float*  cmax   = (float*)(ws + take(N * 4));
  float*  csum   = (float*)(ws + take(N * 4));
  float*  sf1    = (float*)(ws + take(N * 4));
  float*  sf2    = (float*)(ws + take(N * 4));
  float*  mu     = (float*)(ws + take(D * 4));
  float*  var    = (float*)(ws + take(D * 4));
  float*  logits = (float*)(ws + take(B * C * 4));
  // Union region: xg_f/xg_b (84 MB) live only before `last` (105 MB) is written.
  size_t uoff = take((size_t)N * N * 4);
  float*  xgF  = (float*)(ws + uoff);
  float*  xgB  = (float*)(ws + uoff + (size_t)M2 * H4 * 4);
  float*  last = (float*)(ws + uoff);
  if (off > ws_size) return;  // workspace too small; deterministic no-op

  // --- 0: rel matrix, bias sums, bf16 weight copies -------------------------
  k_rel<<<1, 256, 0, stream>>>(tE1, tE2, rel);
  k_bias_sum<<<4, 256, 0, stream>>>(bihF, bhhF, bsumF, H4);
  k_bias_sum<<<4, 256, 0, stream>>>(bihB, bhhB, bsumB, H4);
  k_cvt_bf16<<<(H4 * E + 255) / 256, 256, 0, stream>>>(WihF, WihFb, H4 * E);
  k_cvt_bf16<<<(H4 * E + 255) / 256, 256, 0, stream>>>(WihB, WihBb, H4 * E);
  k_cvt_bf16<<<(H4 * H + 255) / 256, 256, 0, stream>>>(WhhF, WhhFb, H4 * H);
  k_cvt_bf16<<<(H4 * H + 255) / 256, 256, 0, stream>>>(WhhB, WhhBb, H4 * H);
  k_rt_cvt<<<(TH * TH) / 256, 256, 0, stream>>>(Rm, Rtb);

  // --- 1: embedding gather (stacked e1;e2) ----------------------------------
  k_gather<<<(M2 * E) / 256, 256, 0, stream>>>(arg1, arg2, table, estack);

  // --- 2: input-gate GEMMs  xg = estack * Wih^T + (bih+bhh) -----------------
  {
    dim3 grid(H4 / 128, M2 / 64);
    k_gemm_bf16<<<grid, 256, 0, stream>>>(estack, E, WihFb, E, xgF, H4, bsumF, E);
    k_gemm_bf16<<<grid, 256, 0, stream>>>(estack, E, WihBb, E, xgB, H4, bsumB, E);
  }

  // --- 3: LSTM recurrences (fwd then bwd), both sequences stacked M=128 -----
  __bf16* hping[2] = { hbfA, hbfB };
  for (int dir = 0; dir < 2; ++dir) {
    hipMemsetAsync(hbfA, 0, (size_t)2 * B * H * 2, stream);
    hipMemsetAsync(cbuf, 0, (size_t)2 * B * H * 4, stream);
    const __bf16* Whh = dir == 0 ? WhhFb : WhhBb;
    const float*  xg  = dir == 0 ? xgF : xgB;
    for (int s = 0; s < L; ++s) {
      int t = dir == 0 ? s : (L - 1 - s);
      k_lstm_step<<<(2 * B) / 16, 256, 0, stream>>>(
          hping[s & 1], Whh, xg + (size_t)t * H4, cbuf, hping[(s + 1) & 1],
          t1buf, t2buf, t, dir == 0 ? 0 : H);
    }
  }

  // --- 4: attention-style interaction ---------------------------------------
  k_cvt_bf16<<<(N * TH) / 256, 256, 0, stream>>>(t1buf, t1bf, N * TH);
  k_cvt_bf16<<<(N * TH) / 256, 256, 0, stream>>>(t2buf, t2bf, N * TH);
  {
    dim3 grid(TH / 128, N / 64);
    k_gemm_bf16<<<grid, 256, 0, stream>>>(t1bf, TH, Rtb, TH, t1R, TH, nullptr, TH);
  }
  k_cvt_bf16<<<(N * TH) / 256, 256, 0, stream>>>(t1R, t1Rbf, N * TH);
  {
    dim3 grid(N / 256, N / 64);
    k_gemm_last<<<grid, 256, 0, stream>>>(t1Rbf, t2bf, last, rel, TH);
  }

  // --- 5: dual softmax means -------------------------------------------------
  k_rowstat<<<N, 256, 0, stream>>>(last, rmax, rsum);
  k_colstat<<<N, 256, 0, stream>>>(last, cmax, csum);
  k_sf1<<<N, 256, 0, stream>>>(last, rmax, rsum, sf1);
  k_sf2<<<N, 256, 0, stream>>>(last, cmax, csum, sf2);

  // --- 6: BatchNorm stats + fused BN/FC + log_softmax ------------------------
  k_bnstat<<<D / 256, 256, 0, stream>>>(t1buf, t2buf, sf1, sf2, mu, var);
  k_fc<<<B * C, 256, 0, stream>>>(t1buf, t2buf, sf1, sf2, mu, var,
                                  gamma, beta, fcW, fcb, logits);
  k_logsoftmax<<<1, 64, 0, stream>>>(logits, outp);
}
